// NonIterativeModFFN_62380105007319
// MI455X (gfx1250) — matrix operand order repaired and checked by reading, not compiled, not run on hardware
//
#include <hip/hip_runtime.h>
#include <hip/hip_bf16.h>

// ---------------------------------------------------------------------------
// CDNA5 (gfx1250) implementation of the 6-layer gated-FFN "nibble mod" net.
// Per-wave 16-row tiles; projections via v_wmma_f32_16x16x32_f16 (f16 in, fp32
// accumulate). Biases are folded into the otherwise-unused K=16..18 rows of
// the zero-padded B fragments (A carries 1.0 there), so up/gate use the free
// inline-0 C operand. Residual stream lives in fp32 C-fragments; carry
// propagation is register-local (C layout puts position p in VGPR p).
// NOTE: layer 0 only populates hidden rows 0..31 (nib in 0..7), so it needs
// just 2 hidden tiles / 1 down K-chunk despite h=64.
// ---------------------------------------------------------------------------

typedef __attribute__((ext_vector_type(16))) _Float16 v16h;
typedef __attribute__((ext_vector_type(8)))  _Float16 v8h;
typedef __attribute__((ext_vector_type(8)))  float    v8f;

#define S_SCALE 100.0f
#define NLAYER 6
#define NFRAG 28            // sum over layers of (2*ntile + nkchunk)
#define FRAG_HALVES (NFRAG*512)

// Baked weights live in device globals (written by bake kernel every launch;
// deterministic, graph-capture safe: no host memcpy, no mallocs).
__device__ __align__(16) _Float16 g_frags[FRAG_HALVES];

// ---------------- weight generators (transcribed from _bake()) -------------
// Layer 0: only rows 0..31 are non-zero (nib = r>>2 in 0..7, bit = r&3).
__device__ __forceinline__ float wU(int L, int r, int c) {
  switch (L) {
    case 0: if (r < 32) { if (c == 2) return S_SCALE; if (c == 0) return -S_SCALE*100.0f; } return 0.f;
    case 1: if (r < 16 && !(r & 1) && c == 0) return S_SCALE; return 0.f;
    case 2: if (r < 16 && c == 0) return S_SCALE; return 0.f;
    case 3: if (r < 15 && c == 4) return -S_SCALE; return 0.f;
    case 4: if (r < 16 && c == 0) return S_SCALE; return 0.f;
    case 5: if (r < 4 && c == 8) return S_SCALE; return 0.f;
  }
  return 0.f;
}
// bias split into f16-exact pieces: bu = lo + hi40 + mid10
__device__ __forceinline__ float biasLo(int L, int r) {
  switch (L) {
    case 0: if (r < 32) return -S_SCALE*((float)(1 << (r & 3)) - 0.5f); return 0.f;
    case 1: if (r < 16 && !(r & 1)) return -S_SCALE*((float)(r >> 1) - 0.5f); return 0.f;
    case 2: if (r < 16) return -S_SCALE*((float)(r >> 1) - 0.5f); return 0.f;
    case 3: if (r < 15) return -S_SCALE*((float)(r + 1) - 0.5f); return 0.f;
    case 4: if (r < 16) return -S_SCALE*((float)(r >> 1) - 0.5f); return 0.f;
    case 5: return 0.f;
  }
  return 0.f;
}
__device__ __forceinline__ float biasHi40(int L, int r) {   // 40000 * (nib>>2)
  if (L == 0 && r < 32) return ((r >> 4) & 1) ? 40000.f : 0.f;
  return 0.f;
}
__device__ __forceinline__ float biasMid10(int L, int r) {  // 10000 * (nib&3)
  if (L == 0 && r < 32) return 10000.f * (float)((r >> 2) & 3);
  return 0.f;
}
__device__ __forceinline__ float wG(int L, int r, int c) {
  const int slots[4] = {5, 4, 7, 6};  // TEMP, RAW_SUM, CARRY_OUT, CARRY_IN
  switch (L) {
    case 0: if (r < 32 && c == 8) return 1.f; return 0.f;
    case 1: if (r < 16 && !(r & 1) && c == 1) return 1.f; return 0.f;
    case 2: if (r < 16) { if (!(r & 1)) { if (c == 3) return 1.f; }
                          else          { if (c == 2) return -1.f; } } return 0.f;
    case 3: if (r < 15 && c == 8) return 1.f; return 0.f;
    case 4: if (r < 16) { if (!(r & 1)) { if (c == 6) return -1.f; if (c == 4) return 1.f; }
                          else          { if (c == 6) return  1.f; if (c == 3) return 1.f; } } return 0.f;
    case 5: if (r < 4 && c == slots[r]) return -1.f; return 0.f;
  }
  return 0.f;
}
__device__ __forceinline__ float biasG(int L, int r) {
  if (L == 4 && r < 16 && !(r & 1)) return 1.f;
  return 0.f;
}
__device__ __forceinline__ float wD(int L, int j, int r) {   // Wd[j][r]
  const int slots[4] = {5, 4, 7, 6};
  switch (L) {
    case 0: if (j == 7 && r < 32) return (float)(r & 3) / S_SCALE; return 0.f;
    case 1: if (j == 3 && r < 16 && !(r & 1)) return 1.f / S_SCALE; return 0.f;
    case 2: if (j == 4 && r < 16) return 1.f / S_SCALE; return 0.f;
    case 3: if (j == 7 && r < 15) return 1.f / S_SCALE; return 0.f;
    case 4: if (j == 3 && r < 16) return 1.f / S_SCALE; return 0.f;
    case 5: if (r < 4 && j == slots[r]) return 1.f / S_SCALE; return 0.f;
  }
  return 0.f;
}

// -------- bake kernel: write B-fragments in WMMA lane layout ---------------
// Fragment f occupies g_frags[f*512 .. +511]; half index = lane*16 + j.
// B 32x16 f16 layout (wave32): half j of lane half h holds K=8h+j (j<8) or
// K=16+8h+(j-8) (j>=8), column N = lane&15.
// Up/gate fragments carry bias rows at K=16 (lo), K=17 (40000 part),
// K=18 (10000 part); A supplies 1.0 at those K positions.
__global__ void NonIterativeModFFN_bake() {
  const int fb[7] = {0, 5, 10, 15, 20, 25, 28};
  const int heff[6] = {32, 32, 32, 32, 32, 8};   // effective non-zero rows
  const int nt[6] = {2, 2, 2, 2, 2, 1};
  int i = blockIdx.x * blockDim.x + threadIdx.x;
  if (i >= FRAG_HALVES) return;
  int f = i >> 9, r = i & 511, lane = r >> 4, half = r & 15;
  int L = 0;
  while (L < 5 && f >= fb[L + 1]) ++L;
  int k = f - fb[L];
  int kk = (half < 8) ? (lane >> 4) * 8 + half : 16 + (lane >> 4) * 8 + (half - 8);
  float val = 0.f;
  if (k < nt[L]) {                       // up-proj tile k
    int n = k * 16 + (lane & 15);
    if (kk < 16) val = wU(L, n, kk);
    else if (kk == 16) val = biasLo(L, n);
    else if (kk == 17) val = biasHi40(L, n);
    else if (kk == 18) val = biasMid10(L, n);
  } else if (k < 2 * nt[L]) {            // gate-proj tile
    int t = k - nt[L];
    int n = t * 16 + (lane & 15);
    if (kk < 16) val = wG(L, n, kk);
    else if (kk == 16) val = biasG(L, n);
  } else {                               // down-proj K-chunk: B[K=hidden][N=slot]
    int c = k - 2 * nt[L];
    int K = c * 32 + kk;
    if (K < heff[L]) val = wD(L, lane & 15, K);
  }
  g_frags[i] = (_Float16)val;
}

// ------------------------------ main kernel --------------------------------
#define WAVES 8
#define XP 20     // fp32 x tile pitch (dwords)
#define XPH 24    // f16 x tile pitch (halves)
#define HP 40     // f16 hidden tile pitch (halves; max 32 cols used)

__global__ __launch_bounds__(256) void NonIterativeModFFN_62380105007319_kernel(
    const float* __restrict__ xin, float* __restrict__ out, int nTiles) {
  __shared__ __align__(16) float    sXf[WAVES][16 * XP];
  __shared__ __align__(16) _Float16 sXh[WAVES][16 * XPH];
  __shared__ __align__(16) _Float16 sHh[WAVES][16 * HP];

  // compile-time layer metadata -> fully unrolled layer pipeline
  constexpr int NT[NLAYER] = {2, 2, 2, 2, 2, 1};   // 16-col hidden tiles
  constexpr int NK[NLAYER] = {1, 1, 1, 1, 1, 1};   // 32-wide K chunks (down)
  constexpr int FB[NLAYER] = {0, 5, 10, 15, 20, 25};

  const int lane = threadIdx.x & 31;
  const int w    = threadIdx.x >> 5;
  const int col  = lane & 15;          // N index in C layout
  const int hi   = lane >> 4;          // 0: rows 0-7, 1: rows 8-15
  const int colbase = hi ? 8 : 0;      // A-layout K/N split

  float*    Xf = sXf[w];
  _Float16* Xh = sXh[w];
  _Float16* Hh = sHh[w];

  const _Float16 oneK = hi ? (_Float16)0.f : (_Float16)1.f;  // bias-row driver

  const int waveGlobal = (blockIdx.x * (blockDim.x >> 5)) + w;
  const int waveStride = gridDim.x * (blockDim.x >> 5);

  for (int tile = waveGlobal; tile < nTiles; tile += waveStride) {
    const long rowBase = (long)tile * 16;
    // ---- load 16x16 fp32 tile (lane -> row=lane&15, 8 cols from colbase)
    const float4* px =
        (const float4*)(xin + (rowBase + (lane & 15)) * 16 + colbase);
    float4 p0 = px[0], p1 = px[1];
    // stash fp32 tile to LDS to transpose into C layout
    float4* xf = (float4*)(Xf + (lane & 15) * XP + colbase);
    xf[0] = p0; xf[1] = p1;
    // residual stream in WMMA C layout: xc[v] = x[v + 8*hi][col]
    float xc[8];
#pragma unroll
    for (int v = 0; v < 8; ++v) xc[v] = Xf[(v + 8 * hi) * XP + col];

#pragma unroll
    for (int L = 0; L < NLAYER; ++L) {
      const int nt = NT[L], fb = FB[L];
      // ---- refresh f16 mirror of x from xc (C layout -> row-major LDS)
#pragma unroll
      for (int v = 0; v < 8; ++v)
        Xh[(v + 8 * hi) * XPH + col] = (_Float16)xc[v];
      // ---- build A fragment of x: K=0..15 real data, K=16..18 drive biases
      v8h alo = *(const v8h*)(Xh + (lane & 15) * XPH + colbase);
      v16h a;
#pragma unroll
      for (int j = 0; j < 8; ++j) { a[j] = alo[j]; a[j + 8] = (_Float16)0.f; }
      a[8] = oneK; a[9] = oneK; a[10] = oneK;   // K=16,17,18 (lanes 0-15 only)

      // ---- up & gate projections per 16-column hidden tile (C = inline 0)
#pragma unroll
      for (int t = 0; t < NT[L]; ++t) {
        v16h Bu = *(const v16h*)(g_frags + (fb + t) * 512 + lane * 16);
        v16h Bg = *(const v16h*)(g_frags + (fb + nt + t) * 512 + lane * 16);
        v8f u, g;
#pragma unroll
        for (int v = 0; v < 8; ++v) { u[v] = 0.f; g[v] = 0.f; }
        u = __builtin_amdgcn_wmma_f32_16x16x32_f16(false, a, false, Bu,
                                                   (short)0, u, false, false);
        g = __builtin_amdgcn_wmma_f32_16x16x32_f16(false, a, false, Bg,
                                                   (short)0, g, false, false);
#pragma unroll
        for (int v = 0; v < 8; ++v) {
          float hv = fminf(fmaxf(u[v], 0.f), S_SCALE) * g[v];
          Hh[(v + 8 * hi) * HP + t * 16 + col] = (_Float16)hv;
        }
      }

      // ---- down projection: acc = xc + H @ Wd^T (single K=32 chunk/layer)
      v8f acc;
#pragma unroll
      for (int v = 0; v < 8; ++v) acc[v] = xc[v];
#pragma unroll
      for (int c = 0; c < NK[L]; ++c) {
        const int kb = c * 32 + colbase;
        v8h h0 = *(const v8h*)(Hh + (lane & 15) * HP + kb);
        v8h h1 = *(const v8h*)(Hh + (lane & 15) * HP + kb + 16);
        v16h ah;
#pragma unroll
        for (int j = 0; j < 8; ++j) { ah[j] = h0[j]; ah[j + 8] = h1[j]; }
        v16h Bd = *(const v16h*)(g_frags + (fb + 2 * nt + c) * 512 + lane * 16);
        acc = __builtin_amdgcn_wmma_f32_16x16x32_f16(false, ah, false, Bd,
                                                     (short)0, acc, false, false);
      }
#pragma unroll
      for (int v = 0; v < 8; ++v) xc[v] = acc[v];

      // ---- borrow propagation after detect_borrow (layer 3): positions are
      // VGPR-local in C layout (row p lives in xc[p] of the same lane).
      if (L == 3) {
        float co[8], pv[8], ci[8];
        int src = (lane & 16) + 7;  // lane holding column CARRY_OUT for my rows
#pragma unroll
        for (int v = 0; v < 8; ++v) co[v] = __shfl(xc[v], src, 32);
#pragma unroll
        for (int it = 0; it < 7; ++it) {
          pv[0] = 0.f;
#pragma unroll
          for (int v = 1; v < 8; ++v) pv[v] = co[v - 1];
#pragma unroll
          for (int v = 0; v < 8; ++v) {
            ci[v] = pv[v];
            co[v] = fminf(fmaxf(co[v] + pv[v], 0.f), 1.f);
          }
        }
#pragma unroll
        for (int v = 0; v < 8; ++v)
          xc[v] = (col == 6) ? ci[v] : ((col == 7) ? co[v] : xc[v]);
      }
    }

    // ---- store result (coalesced: 64B per half-wave per VGPR)
#pragma unroll
    for (int v = 0; v < 8; ++v)
      out[(rowBase + v + 8 * hi) * 16 + col] = xc[v];
  }
}

extern "C" void kernel_launch(void* const* d_in, const int* in_sizes, int n_in,
                              void* d_out, int out_size, void* d_ws, size_t ws_size,
                              hipStream_t stream) {
  (void)n_in; (void)d_ws; (void)ws_size; (void)out_size;
  const float* xin = (const float*)d_in[0];
  float* out = (float*)d_out;
  const int nTiles = in_sizes[0] / 256;   // 16 rows x 16 slots per tile

  NonIterativeModFFN_bake<<<(FRAG_HALVES + 255) / 256, 256, 0, stream>>>();

  int blocks = (nTiles + WAVES - 1) / WAVES;
  if (blocks > 2048) blocks = 2048;
  if (blocks < 1) blocks = 1;
  NonIterativeModFFN_62380105007319_kernel<<<blocks, 256, 0, stream>>>(xin, out, nTiles);
}
